// KNO_DARCY_PWC_19224273617248
// MI455X (gfx1250) — compile-verified, matrix operand order checked by reading, not run on hardware
//
#include <hip/hip_runtime.h>
#include <hip/hip_bf16.h>

#define Q 128
#define NPIX (Q * Q)   // 16384
#define LIFT 64
#define DEPTH 4

typedef _Float16 half16 __attribute__((ext_vector_type(16)));
typedef float    float8 __attribute__((ext_vector_type(8)));

union Frag {
    half16 v;
    _Float16 h[16];
    unsigned int u[8];
};

// Branch-free tanh-approx GELU.
// 0.5*x*(1+tanh(z)) == x * sigmoid(2z) == x * rcp(1 + exp2(-2*log2(e)*z))
__device__ __forceinline__ float gelu_tanh(float x) {
    const float c0 = 0.7978845608028654f;   // sqrt(2/pi)
    const float c1 = 0.044715f;
    float z = c0 * (x + c1 * x * x * x);
#if __has_builtin(__builtin_amdgcn_tanhf)
    return 0.5f * x * (1.0f + __builtin_amdgcn_tanhf(z));   // v_tanh_f32 (TRANS)
#else
    float e = __builtin_amdgcn_exp2f(-2.885390081777927f * z);  // exp(-2z)
    return x * __builtin_amdgcn_rcpf(1.0f + e);
#endif
}

// 16-bit A/B fragment element -> K index mapping (ISA 7.12.2):
// lanes 0-15 hold K {0..7,16..23}, lanes 16-31 hold K {8..15,24..31}
__device__ __forceinline__ int kmap(int e, int hi) {
    return (e & 7) + ((e >> 3) << 4) + (hi << 3);
}
// dword-pair variant: pair t covers elements (2t, 2t+1) -> K (k, k+1)
__device__ __forceinline__ int kpair(int t, int hi) {
    return 2 * (t & 3) + ((t >> 2) << 4) + (hi << 3);
}

__device__ __forceinline__ float8 wmma16(const Frag& a, const Frag& b, float8 c) {
    return __builtin_amdgcn_wmma_f32_16x16x32_f16(false, a.v, false, b.v,
                                                  (short)0, c, false, false);
}

#if __has_builtin(__builtin_amdgcn_global_load_async_to_lds_b128) && \
    __has_builtin(__builtin_amdgcn_s_wait_asynccnt)
#define HAVE_ASYNC_LDS 1
typedef int v4i __attribute__((vector_size(16)));
typedef __attribute__((address_space(1))) v4i gv4i;   // global (prints as __device__)
typedef __attribute__((address_space(3))) v4i sv4i;   // LDS    (prints as __shared__)
#else
#define HAVE_ASYNC_LDS 0
#endif

// ---------------------------------------------------------------- lift
__global__ void __launch_bounds__(256)
lift_kernel(const float* __restrict__ fx, const float* __restrict__ xg,
            const float* __restrict__ Wl, const float* __restrict__ bl,
            float* __restrict__ f0) {
    int idx = blockIdx.x * 256 + threadIdx.x;   // p*64 + l
    int p = idx >> 6, l = idx & 63;
    float a = fx[p], b = xg[p * 2], c = xg[p * 2 + 1];
    f0[idx] = a * Wl[l * 3] + b * Wl[l * 3 + 1] + c * Wl[l * 3 + 2] + bl[l];
}

// ------------------------------------------------- f32 -> f16 (two layouts)
__global__ void __launch_bounds__(256)
pack_kernel(const float* __restrict__ f, _Float16* __restrict__ fp,
            _Float16* __restrict__ fc) {
    int idx = blockIdx.x * 256 + threadIdx.x;   // p*64 + c
    _Float16 h = (_Float16)f[idx];
    fp[idx] = h;                                // row-major (p, c)
    fc[(idx & 63) * NPIX + (idx >> 6)] = h;     // channel-major (c, p)
}

// ------------------------------------------- kernel-matrix generation (G1,G2T)
__global__ void __launch_bounds__(256)
ggen_kernel(const float* __restrict__ kW1, const float* __restrict__ kb1,
            const float* __restrict__ kW2, const float* __restrict__ kb2,
            const float* __restrict__ kW3, const float* __restrict__ kb3,
            const float* __restrict__ qw,
            _Float16* __restrict__ G1, _Float16* __restrict__ G2T, int depth) {
    __shared__ _Float16 w2s[64 * 64];   // W2[c] row-major (g,h) as f16
    __shared__ float w1a[64], w1b[64], b1s[64], b2s[64], w3s[64];

    const int c = blockIdx.z;
    const int dir = blockIdx.y;
    const int tid = threadIdx.x;
    const int dc = (depth * 2 + dir) * 64 + c;

    const float* W2 = kW2 + (size_t)dc * 64 * 64;
#pragma unroll
    for (int i = 0; i < 16; ++i) {
        int idx = tid * 16 + i;
        w2s[idx] = (_Float16)W2[idx];
    }
    if (tid < 64) {
        w1a[tid] = kW1[dc * 128 + tid * 2 + 0];
        w1b[tid] = kW1[dc * 128 + tid * 2 + 1];
        b1s[tid] = kb1[dc * 64 + tid];
        b2s[tid] = kb2[dc * 64 + tid];
        w3s[tid] = kW3[dc * 64 + tid];
    }
    const float b3 = kb3[dc];
    __syncthreads();

    const int lane = tid & 31;
    const int wave = tid >> 5;
    const int m = lane & 15;
    const int hi = lane >> 4;

    // B[k=h][n=g] = W2[g][h]; w2s is [g*64+h] so K-pairs are contiguous dwords
    Frag B[4][2];
#pragma unroll
    for (int nt = 0; nt < 4; ++nt) {
        int n = nt * 16 + m;
#pragma unroll
        for (int kh = 0; kh < 2; ++kh)
#pragma unroll
            for (int t = 0; t < 8; ++t) {
                int k = kh * 32 + kpair(t, hi);
                B[nt][kh].u[t] = *(const unsigned int*)&w2s[n * 64 + k];
            }
    }

    const float inv127 = 1.0f / 127.0f;

    for (int tt = 0; tt < 8; ++tt) {
        int tile = blockIdx.x * 64 + wave * 8 + tt;   // 16-pixel tile
        int p = tile * 16 + m;
        float xi = (float)(p >> 7) * inv127;
        float xj = (float)(p & 127) * inv127;

        // h1 = gelu(pairs @ W1^T + b1) directly in A-fragment layout
        Frag A0, A1;
#pragma unroll
        for (int e = 0; e < 16; ++e) {
            int k = kmap(e, hi);
            A0.h[e] = (_Float16)gelu_tanh(xi * w1a[k] + xj * w1b[k] + b1s[k]);
            int k2 = k + 32;
            A1.h[e] = (_Float16)gelu_tanh(xi * w1a[k2] + xj * w1b[k2] + b1s[k2]);
        }

        float sum[8];
#pragma unroll
        for (int r = 0; r < 8; ++r) sum[r] = 0.0f;

#pragma unroll
        for (int nt = 0; nt < 4; ++nt) {
            float8 acc = {};
            acc = wmma16(A0, B[nt][0], acc);
            acc = wmma16(A1, B[nt][1], acc);
            int g = nt * 16 + m;           // C-layout: col = lane&15 (+nt*16)
            float b2g = b2s[g], w3g = w3s[g];
#pragma unroll
            for (int r = 0; r < 8; ++r)
                sum[r] += gelu_tanh(acc[r] + b2g) * w3g;
        }
        // reduce over g (lanes within each 16-lane half)
#pragma unroll
        for (int r = 0; r < 8; ++r) {
            float s = sum[r];
            s += __shfl_xor(s, 1, 32);
            s += __shfl_xor(s, 2, 32);
            s += __shfl_xor(s, 4, 32);
            s += __shfl_xor(s, 8, 32);
            sum[r] = s;
        }
        if (m == 0) {                       // lanes 0 (rows 0-7) and 16 (rows 8-15)
#pragma unroll
            for (int r = 0; r < 8; ++r) {
                int p2 = tile * 16 + r + 8 * hi;
                int y2 = p2 & 127;
                float val = (sum[r] + b3) * qw[y2];          // G * w[y]
                if (dir == 0)
                    G1[c * NPIX + p2] = (_Float16)val;        // row-major (i,k)
                else
                    G2T[c * NPIX + y2 * 128 + (p2 >> 7)] = (_Float16)val; // (j,k)
            }
        }
    }
}

// ------------------------------------------------------------- skip GEMM
__global__ void __launch_bounds__(256)
skip_kernel(const _Float16* __restrict__ fh_p, const float* __restrict__ pwW,
            const float* __restrict__ pwb, float* __restrict__ S, int depth) {
    __shared__ _Float16 wts[64 * 64];    // [m][l] = pw_W[depth][m][l]
    const int tid = threadIdx.x;
    const float* W = pwW + (size_t)depth * 64 * 64;
#pragma unroll
    for (int i = 0; i < 16; ++i) {
        int idx = tid * 16 + i;
        wts[idx] = (_Float16)W[idx];
    }
    __syncthreads();

    const int lane = tid & 31, wave = tid >> 5;
    const int m = lane & 15, hi = lane >> 4;
    const int pt = blockIdx.x * 8 + wave;

    Frag A[2];
#pragma unroll
    for (int kh = 0; kh < 2; ++kh)
#pragma unroll
        for (int t = 0; t < 8; ++t) {
            int k = kh * 32 + kpair(t, hi);
            A[kh].u[t] = *(const unsigned int*)&fh_p[(pt * 16 + m) * 64 + k];
        }

#pragma unroll
    for (int nt = 0; nt < 4; ++nt) {
        int n = nt * 16 + m;
        Frag B0, B1;
#pragma unroll
        for (int t = 0; t < 8; ++t) {
            B0.u[t] = *(const unsigned int*)&wts[n * 64 + kpair(t, hi)];
            B1.u[t] = *(const unsigned int*)&wts[n * 64 + 32 + kpair(t, hi)];
        }
        float8 acc = {};
        acc = wmma16(A[0], B0, acc);
        acc = wmma16(A[1], B1, acc);
        float bn = pwb[depth * 64 + n];
#pragma unroll
        for (int r = 0; r < 8; ++r) {
            int p2 = pt * 16 + r + 8 * hi;
            S[p2 * 64 + n] = acc[r] + bn;
        }
    }
}

// -------------------------------------- integral-operator apply (per channel)
__global__ void __launch_bounds__(256)
apply_kernel(const _Float16* __restrict__ G1, const _Float16* __restrict__ G2T,
             const _Float16* __restrict__ fh_c, const float* __restrict__ S,
             float* __restrict__ fout, int act) {
    extern __shared__ __align__(16) _Float16 smem[];
    _Float16* sG1 = smem;             // 128x128 f16
    _Float16* sF  = smem + NPIX;      // 128x128 f16
    _Float16* sG2 = smem + 2 * NPIX;  // 128x128 f16 (already transposed)

    const int c = blockIdx.x;
    const int tid = threadIdx.x;

#if HAVE_ASYNC_LDS
    {
        // CDNA5 async DMA: global -> LDS without touching VGPRs (ASYNCcnt)
        const _Float16* srcs[3] = { G1 + (size_t)c * NPIX,
                                    fh_c + (size_t)c * NPIX,
                                    G2T + (size_t)c * NPIX };
        _Float16* dsts[3] = { sG1, sF, sG2 };
#pragma unroll
        for (int a = 0; a < 3; ++a) {
#pragma unroll
            for (int i = 0; i < 8; ++i) {
                int e8 = (tid + 256 * i) * 8;   // 8 halves (16B) per transfer
                __builtin_amdgcn_global_load_async_to_lds_b128(
                    (gv4i*)(srcs[a] + e8), (sv4i*)(dsts[a] + e8), 0, 0);
            }
        }
        __builtin_amdgcn_s_wait_asynccnt(0);
    }
    __syncthreads();
#else
    {
        const uint4* g1 = (const uint4*)(G1 + (size_t)c * NPIX);
        const uint4* gf = (const uint4*)(fh_c + (size_t)c * NPIX);
        const uint4* g2 = (const uint4*)(G2T + (size_t)c * NPIX);
        uint4* t1 = (uint4*)sG1; uint4* t2 = (uint4*)sF; uint4* t3 = (uint4*)sG2;
#pragma unroll
        for (int i = 0; i < 8; ++i) {
            int idx = tid + 256 * i;      // 2048 uint4 per matrix
            t1[idx] = g1[idx];
            t2[idx] = gf[idx];
            t3[idx] = g2[idx];
        }
    }
    __syncthreads();
#endif

    const int lane = tid & 31, wave = tid >> 5;
    const int m = lane & 15, hi = lane >> 4;
    const int row = wave * 16 + m;    // wave owns 16-row stripe

    Frag A1[4], A2[4];
#pragma unroll
    for (int kb = 0; kb < 4; ++kb)
#pragma unroll
        for (int t = 0; t < 8; ++t) {
            int k = kb * 32 + kpair(t, hi);
            A1[kb].u[t] = *(const unsigned int*)&sG1[row * 128 + k];
            A2[kb].u[t] = *(const unsigned int*)&sF[row * 128 + k];
        }

    for (int nt = 0; nt < 8; ++nt) {
        int n = nt * 16 + m;
        float8 acc = {};
#pragma unroll
        for (int kb = 0; kb < 4; ++kb) {
            Frag B1, B2;
#pragma unroll
            for (int e = 0; e < 16; ++e) {
                int k = kb * 32 + kmap(e, hi);
                B1.h[e] = sF[k * 128 + n];
                B2.h[e] = sG2[k * 128 + n];
            }
            acc = wmma16(A1[kb], B1, acc);   // G1 @ F
            acc = wmma16(A2[kb], B2, acc);   // F @ G2^T
        }
#pragma unroll
        for (int r = 0; r < 8; ++r) {
            int p = (wave * 16 + r + 8 * hi) * 128 + n;
            float val = S[p * 64 + c] + acc[r];
            if (act) val = gelu_tanh(val);
            fout[p * 64 + c] = val;
        }
    }
}

// ---------------------------------------------------------------- head MLP
__global__ void __launch_bounds__(256)
head_kernel(const _Float16* __restrict__ fh_p,
            const float* __restrict__ P1, const float* __restrict__ pb1,
            const float* __restrict__ P2, const float* __restrict__ pb2,
            const float* __restrict__ P3, const float* __restrict__ pb3,
            float* __restrict__ out) {
    __shared__ _Float16 s1[64 * 64];
    __shared__ _Float16 s2[64 * 64];
    __shared__ _Float16 r1s[128 * 64];

    const int tid = threadIdx.x;
#pragma unroll
    for (int i = 0; i < 16; ++i) {
        int idx = tid * 16 + i;
        s1[idx] = (_Float16)P1[idx];
        s2[idx] = (_Float16)P2[idx];
    }
    __syncthreads();

    const int lane = tid & 31, wave = tid >> 5;
    const int m = lane & 15, hi = lane >> 4;
    const int pt = blockIdx.x * 8 + wave;

    // stage 1: r1 = gelu(f @ P1^T + pb1)
    Frag A[2];
#pragma unroll
    for (int kh = 0; kh < 2; ++kh)
#pragma unroll
        for (int t = 0; t < 8; ++t) {
            int k = kh * 32 + kpair(t, hi);
            A[kh].u[t] = *(const unsigned int*)&fh_p[(pt * 16 + m) * 64 + k];
        }
#pragma unroll
    for (int nt = 0; nt < 4; ++nt) {
        int n = nt * 16 + m;
        Frag B0, B1;
#pragma unroll
        for (int t = 0; t < 8; ++t) {
            B0.u[t] = *(const unsigned int*)&s1[n * 64 + kpair(t, hi)];
            B1.u[t] = *(const unsigned int*)&s1[n * 64 + 32 + kpair(t, hi)];
        }
        float8 acc = {};
        acc = wmma16(A[0], B0, acc);
        acc = wmma16(A[1], B1, acc);
        float bn = pb1[n];
#pragma unroll
        for (int r = 0; r < 8; ++r) {
            int lrow = wave * 16 + r + 8 * hi;
            r1s[lrow * 64 + n] = (_Float16)gelu_tanh(acc[r] + bn);
        }
    }
    __syncthreads();

    // stage 2 + 3: out = gelu(r1 @ P2^T + pb2) @ P3^T + pb3
    Frag A2f[2];
#pragma unroll
    for (int kh = 0; kh < 2; ++kh)
#pragma unroll
        for (int t = 0; t < 8; ++t) {
            int k = kh * 32 + kpair(t, hi);
            A2f[kh].u[t] = *(const unsigned int*)&r1s[(wave * 16 + m) * 64 + k];
        }
    float sum[8];
#pragma unroll
    for (int r = 0; r < 8; ++r) sum[r] = 0.0f;
#pragma unroll
    for (int nt = 0; nt < 4; ++nt) {
        int n = nt * 16 + m;
        Frag B0, B1;
#pragma unroll
        for (int t = 0; t < 8; ++t) {
            B0.u[t] = *(const unsigned int*)&s2[n * 64 + kpair(t, hi)];
            B1.u[t] = *(const unsigned int*)&s2[n * 64 + 32 + kpair(t, hi)];
        }
        float8 acc = {};
        acc = wmma16(A2f[0], B0, acc);
        acc = wmma16(A2f[1], B1, acc);
        float bn = pb2[n], w3n = P3[n];
#pragma unroll
        for (int r = 0; r < 8; ++r)
            sum[r] += gelu_tanh(acc[r] + bn) * w3n;
    }
#pragma unroll
    for (int r = 0; r < 8; ++r) {
        float s = sum[r];
        s += __shfl_xor(s, 1, 32);
        s += __shfl_xor(s, 2, 32);
        s += __shfl_xor(s, 4, 32);
        s += __shfl_xor(s, 8, 32);
        if (m == 0) out[pt * 16 + r + 8 * hi] = s + pb3[0];
    }
}

// ------------------------------------------------------------------ driver
extern "C" void kernel_launch(void* const* d_in, const int* in_sizes, int n_in,
                              void* d_out, int out_size, void* d_ws, size_t ws_size,
                              hipStream_t stream) {
    (void)in_sizes; (void)n_in; (void)out_size; (void)ws_size;
    const float* f_x   = (const float*)d_in[0];
    const float* xgrid = (const float*)d_in[1];
    const float* qw    = (const float*)d_in[2];
    const float* Wl    = (const float*)d_in[3];
    const float* bl    = (const float*)d_in[4];
    const float* pwW   = (const float*)d_in[5];
    const float* pwb   = (const float*)d_in[6];
    const float* kW1   = (const float*)d_in[7];
    const float* kb1   = (const float*)d_in[8];
    const float* kW2   = (const float*)d_in[9];
    const float* kb2   = (const float*)d_in[10];
    const float* kW3   = (const float*)d_in[11];
    const float* kb3   = (const float*)d_in[12];
    const float* P1    = (const float*)d_in[13];
    const float* pb1   = (const float*)d_in[14];
    const float* P2    = (const float*)d_in[15];
    const float* pb2   = (const float*)d_in[16];
    const float* P3    = (const float*)d_in[17];
    const float* pb3   = (const float*)d_in[18];
    float* out = (float*)d_out;

    char* ws = (char*)d_ws;
    float* fA = (float*)ws;                           // 4 MB  current f
    float* fB = (float*)(ws + (4 << 20));             // 4 MB  next f
    float* S  = (float*)(ws + (8 << 20));             // 4 MB  skip
    _Float16* fh_p = (_Float16*)(ws + (12 << 20));    // 2 MB  f16 (p,c)
    _Float16* fh_c = (_Float16*)(ws + (14 << 20));    // 2 MB  f16 (c,p)
    _Float16* G1   = (_Float16*)(ws + (16 << 20));    // 2 MB
    _Float16* G2T  = (_Float16*)(ws + (18 << 20));    // 2 MB

    lift_kernel<<<NPIX * 64 / 256, 256, 0, stream>>>(f_x, xgrid, Wl, bl, fA);

    for (int d = 0; d < DEPTH; ++d) {
        pack_kernel<<<NPIX * 64 / 256, 256, 0, stream>>>(fA, fh_p, fh_c);
        ggen_kernel<<<dim3(16, 2, 64), 256, 0, stream>>>(kW1, kb1, kW2, kb2,
                                                         kW3, kb3, qw, G1, G2T, d);
        skip_kernel<<<128, 256, 0, stream>>>(fh_p, pwW, pwb, S, d);
        apply_kernel<<<64, 256, 3 * NPIX * sizeof(_Float16), stream>>>(
            G1, G2T, fh_c, S, fB, (d < DEPTH - 1) ? 1 : 0);
        float* t = fA; fA = fB; fB = t;
    }
    pack_kernel<<<NPIX * 64 / 256, 256, 0, stream>>>(fA, fh_p, fh_c);
    head_kernel<<<128, 256, 0, stream>>>(fh_p, P1, pb1, P2, pb2, P3, pb3, out);
}
